// SNNLoss_32968168964800
// MI455X (gfx1250) — matrix operand order repaired and checked by reading, not compiled, not run on hardware
//
#include <hip/hip_runtime.h>
#include <hip/hip_bf16.h>
#include <math.h>

// ---- CDNA5 vector types for WMMA ----
typedef __bf16 bf16;
typedef bf16  v16bf __attribute__((ext_vector_type(16)));
typedef bf16  v8bf  __attribute__((ext_vector_type(8)));
typedef float v8f   __attribute__((ext_vector_type(8)));

#define BATCH 8192
#define DIM   1024
#define BM 128
#define BN 128
#define BK 32
#define COLSPLITS 8
#define CTS_PER_BLOCK ((BATCH / COLSPLITS) / BN)   // 8 column tiles per block
#define ROWTILES (BATCH / BM)                      // 64
#define LDS_STRIDE 40   // halfwords per staged row (80B): 16B-aligned, bank-conflict free
#define STG_ELEMS (BM * LDS_STRIDE)                // 5120 bf16 per staged matrix

// ---- workspace layout (bytes) ----
#define XHI_OFF   ((size_t)0)
#define XLO_OFF   ((size_t)BATCH * DIM * 2)
#define XN_OFF    ((size_t)BATCH * DIM * 4)
#define STATS_OFF (XN_OFF + (size_t)BATCH * 4)
#define ACC_OFF   (STATS_OFF + (size_t)BATCH * COLSPLITS * 16)

// gfx1250 async global->LDS copy (ASYNCcnt path, no VGPR round-trip).
// GVS addressing: 64-bit SGPR base + 32-bit VGPR byte offset. VDST carries the
// per-lane LDS byte address (low 32 bits of the generic shared pointer).
#define ASYNC_LD16(ldsp, base, off)                                           \
  asm volatile("global_load_async_to_lds_b128 %0, %1, %2"                     \
               :: "v"((unsigned)(unsigned long long)(ldsp)),                  \
                  "v"((unsigned)(off)),                                       \
                  "s"((unsigned long long)(base)) : "memory")
#define ASYNC_WAIT0() asm volatile("s_wait_asynccnt 0x0" ::: "memory")

// =====================================================================
// Kernel 1: split x into bf16 hi/lo, compute row norms, zero accumulator
// =====================================================================
__global__ __launch_bounds__(256) void snnl_prep(const float* __restrict__ x,
                                                 bf16* __restrict__ xhi,
                                                 bf16* __restrict__ xlo,
                                                 float* __restrict__ xn,
                                                 float* __restrict__ acc) {
  const int r = blockIdx.x;
  const int t = threadIdx.x;
  const float* row = x + (size_t)r * DIM;
  bf16* hrow = xhi + (size_t)r * DIM;
  bf16* lrow = xlo + (size_t)r * DIM;
  float p = 0.f;
#pragma unroll
  for (int i = 0; i < DIM / 256; ++i) {
    const int k = t + i * 256;
    const float v = row[k];
    const bf16 h = (bf16)v;
    const bf16 l = (bf16)(v - (float)h);
    hrow[k] = h;
    lrow[k] = l;
    p += v * v;
  }
  __shared__ float red[256];
  red[t] = p;
  __syncthreads();
  for (int s = 128; s > 0; s >>= 1) {
    if (t < s) red[t] += red[t + s];
    __syncthreads();
  }
  if (t == 0) {
    xn[r] = red[0];
    if (r == 0) *acc = 0.f;
  }
}

// =====================================================================
// Kernel 2: fused Gram-tile GEMM (bf16x3 WMMA) + online logsumexp
// grid = (64 row tiles, 8 column splits), block = 256 threads (8 waves)
// Double-buffered async global->LDS staging overlapped with WMMA.
// =====================================================================
__global__ __launch_bounds__(256) void snnl_main(const bf16* __restrict__ xhi,
                                                 const bf16* __restrict__ xlo,
                                                 const float* __restrict__ xn,
                                                 const int* __restrict__ y,
                                                 const int* __restrict__ Tptr,
                                                 float4* __restrict__ stats) {
  const int t    = threadIdx.x;
  const int lane = t & 31;
  const int wave = t >> 5;       // 0..7
  const int rw   = wave >> 2;    // 0..1 row-wave
  const int cw   = wave & 3;     // 0..3 col-wave
  const int half = lane >> 4;    // 0/1 half of wave32
  const int ln16 = lane & 15;

  const int r0    = blockIdx.x * BM;
  const int cs    = blockIdx.y;
  const int cbase = cs * (BATCH / COLSPLITS);

  const float T = (float)(*Tptr);

  // Double-buffered staging (2 x 4 matrices: Ah, Al, Bh, Bl) overlaid with
  // the Gram tile used by the epilogue (live at disjoint times).
  __shared__ union __align__(16) SH {
    bf16  stg[2][4][STG_ELEMS];         // 80 KB
    float gt[BM * 65];                  // 33 KB (half of BN columns per phase)
  } sh;
  __shared__ float xnc[BN];
  __shared__ int   yc[BN];

  // Row-owner state (threads 0..127 each own one anchor row of this block).
  float xi = 0.f;
  int   yi = 0;
  float mden = -1e30f, sden = 0.f, mnum = -1e30f, snum = 0.f;
  if (t < BM) {
    xi = xn[r0 + t];
    yi = y[r0 + t];
  }

  const int wm0   = rw * 64;      // wave row origin in tile
  const int wn0   = cw * 32;      // wave col origin in tile
  const int koffA = half * 8;     // A frag: halfword offset of first b128 chunk
  const int koffB = half * 8;     // B frag: same k order as A frag

  for (int ct = 0; ct < CTS_PER_BLOCK; ++ct) {
    const int c0 = cbase + ct * BN;
    __syncthreads();   // previous epilogue done before reusing xnc/yc/stage
    if (t < BN) {
      xnc[t] = xn[c0 + t];
      yc[t]  = y[c0 + t];
    }

    // Issue 8 async 16B copies per thread: K chunk for A rows and B rows.
    auto stage_async = [&](int ks, int buf) {
      const int kb = ks * BK;
#pragma unroll
      for (int it = 0; it < 2; ++it) {
        const int idx = t + it * 256;          // 0..511 -> (row, 16B sub-chunk)
        const int row = idx >> 2;
        const int sub = idx & 3;
        const int lo  = row * LDS_STRIDE + sub * 8;
        const unsigned gA = (unsigned)(((r0 + row) * DIM + kb + sub * 8) * 2);
        const unsigned gB = (unsigned)(((c0 + row) * DIM + kb + sub * 8) * 2);
        ASYNC_LD16(&sh.stg[buf][0][lo], xhi, gA);
        ASYNC_LD16(&sh.stg[buf][1][lo], xlo, gA);
        ASYNC_LD16(&sh.stg[buf][2][lo], xhi, gB);
        ASYNC_LD16(&sh.stg[buf][3][lo], xlo, gB);
      }
    };

    v8f acc[4][2];
#pragma unroll
    for (int mt = 0; mt < 4; ++mt)
#pragma unroll
      for (int nt = 0; nt < 2; ++nt) acc[mt][nt] = v8f{};

    // Pipeline prologue: fill buffer 0.
    stage_async(0, 0);
    ASYNC_WAIT0();
    __syncthreads();

    for (int ks = 0; ks < DIM / BK; ++ks) {
      const int buf = ks & 1;
      // Prefetch next K chunk into the other buffer while we compute.
      if (ks + 1 < DIM / BK) stage_async(ks + 1, buf ^ 1);

      const bf16* sAh = sh.stg[buf][0];
      const bf16* sAl = sh.stg[buf][1];
      const bf16* sBh = sh.stg[buf][2];
      const bf16* sBl = sh.stg[buf][3];

      // ---- A fragments: 16x32 bf16, per ISA layout ----
      v16bf ah[4], al[4], bh[2], bl[2];
#pragma unroll
      for (int mt = 0; mt < 4; ++mt) {
        const int rrow = (wm0 + mt * 16 + ln16) * LDS_STRIDE;
        v8bf a0 = *(const v8bf*)&sAh[rrow + koffA];
        v8bf a1 = *(const v8bf*)&sAh[rrow + koffA + 16];
        ah[mt] = __builtin_shufflevector(a0, a1, 0, 1, 2, 3, 4, 5, 6, 7, 8, 9,
                                         10, 11, 12, 13, 14, 15);
        v8bf c0v = *(const v8bf*)&sAl[rrow + koffA];
        v8bf c1v = *(const v8bf*)&sAl[rrow + koffA + 16];
        al[mt] = __builtin_shufflevector(c0v, c1v, 0, 1, 2, 3, 4, 5, 6, 7, 8, 9,
                                         10, 11, 12, 13, 14, 15);
      }
      // ---- B fragments: 32x16 bf16 (cols = rows of x; Gram symmetry) ----
#pragma unroll
      for (int nt = 0; nt < 2; ++nt) {
        const int brow = (wn0 + nt * 16 + ln16) * LDS_STRIDE;
        v8bf b0 = *(const v8bf*)&sBh[brow + koffB];
        v8bf b1 = *(const v8bf*)&sBh[brow + koffB + 16];
        bh[nt] = __builtin_shufflevector(b0, b1, 0, 1, 2, 3, 4, 5, 6, 7, 8, 9,
                                         10, 11, 12, 13, 14, 15);
        v8bf d0 = *(const v8bf*)&sBl[brow + koffB];
        v8bf d1 = *(const v8bf*)&sBl[brow + koffB + 16];
        bl[nt] = __builtin_shufflevector(d0, d1, 0, 1, 2, 3, 4, 5, 6, 7, 8, 9,
                                         10, 11, 12, 13, 14, 15);
      }
      // ---- bf16x3: hi*hi + hi*lo + lo*hi, fp32 accumulate ----
#pragma unroll
      for (int mt = 0; mt < 4; ++mt)
#pragma unroll
        for (int nt = 0; nt < 2; ++nt) {
          acc[mt][nt] = __builtin_amdgcn_wmma_f32_16x16x32_bf16(
              false, ah[mt], false, bh[nt], (short)0, acc[mt][nt], false, false);
          acc[mt][nt] = __builtin_amdgcn_wmma_f32_16x16x32_bf16(
              false, ah[mt], false, bl[nt], (short)0, acc[mt][nt], false, false);
          acc[mt][nt] = __builtin_amdgcn_wmma_f32_16x16x32_bf16(
              false, al[mt], false, bh[nt], (short)0, acc[mt][nt], false, false);
        }

      // Gate next iteration: our async copies done, then block-wide visible.
      ASYNC_WAIT0();
      __syncthreads();
    }  // ks

    // ---- epilogue in two 64-column phases (gt overlays the stage bufs) ----
    for (int ph = 0; ph < 2; ++ph) {
      __syncthreads();
      if ((cw >> 1) == ph) {
#pragma unroll
        for (int mt = 0; mt < 4; ++mt)
#pragma unroll
          for (int nt = 0; nt < 2; ++nt)
#pragma unroll
            for (int j = 0; j < 8; ++j) {
              const int m  = wm0 + mt * 16 + half * 8 + j;
              const int nl = (wn0 - ph * 64) + nt * 16 + ln16;
              sh.gt[m * 65 + nl] = acc[mt][nt][j];
            }
      }
      __syncthreads();
      if (t < BM) {
        const int gi = r0 + t;
#pragma unroll 4
        for (int n = 0; n < 64; ++n) {
          const int col = ph * 64 + n;
          const int gj  = c0 + col;
          if (gj == gi) continue;  // exclude diagonal from num and den
          const float g  = sh.gt[t * 65 + n];
          float d2 = xi + xnc[col] - 2.f * g;
          d2 = d2 > 0.f ? d2 : 0.f;
          const float e = -sqrtf(d2) * T;
          // online logsumexp: denominator (all j != i)
          if (e > mden) {
            sden = sden * __expf(mden - e) + 1.f;
            mden = e;
          } else {
            sden += __expf(e - mden);
          }
          // numerator (same label, j != i)
          if (yc[col] == yi) {
            if (e > mnum) {
              snum = snum * __expf(mnum - e) + 1.f;
              mnum = e;
            } else {
              snum += __expf(e - mnum);
            }
          }
        }
      }
    }
  }  // ct

  if (t < BM) {
    stats[(size_t)(r0 + t) * COLSPLITS + cs] = make_float4(mnum, snum, mden, sden);
  }
}

// =====================================================================
// Kernel 3: merge per-split logsumexp partials, reduce loss sum
// =====================================================================
__global__ __launch_bounds__(256) void snnl_reduce(const float4* __restrict__ stats,
                                                   float* __restrict__ acc) {
  const int i = blockIdx.x * blockDim.x + threadIdx.x;  // row 0..8191
  float mn = -1e30f, sn = 0.f, md = -1e30f, sd = 0.f;
#pragma unroll
  for (int p = 0; p < COLSPLITS; ++p) {
    const float4 s = stats[(size_t)i * COLSPLITS + p];
    if (s.y > 0.f) {
      if (s.x > mn) { sn = sn * __expf(mn - s.x) + s.y; mn = s.x; }
      else          { sn += s.y * __expf(s.x - mn); }
    }
    if (s.w > 0.f) {
      if (s.z > md) { sd = sd * __expf(md - s.z) + s.w; md = s.z; }
      else          { sd += s.w * __expf(s.z - md); }
    }
  }
  const float num = (sn > 0.f) ? (mn + __logf(sn)) : 0.f;  // isinf(num) fixup
  const float den = md + __logf(sd);
  float val = num - den;

  __shared__ float red[256];
  red[threadIdx.x] = val;
  __syncthreads();
  for (int s = 128; s > 0; s >>= 1) {
    if (threadIdx.x < s) red[threadIdx.x] += red[threadIdx.x + s];
    __syncthreads();
  }
  if (threadIdx.x == 0) atomicAdd(acc, red[0]);
}

// =====================================================================
// Kernel 4: final scale
// =====================================================================
__global__ void snnl_finalize(const float* __restrict__ acc, float* __restrict__ out) {
  out[0] = -acc[0] / (float)BATCH;
}

// =====================================================================
extern "C" void kernel_launch(void* const* d_in, const int* in_sizes, int n_in,
                              void* d_out, int out_size, void* d_ws, size_t ws_size,
                              hipStream_t stream) {
  (void)in_sizes; (void)n_in; (void)out_size; (void)ws_size;
  const float* x  = (const float*)d_in[0];
  const int*   y  = (const int*)d_in[1];
  const int*   Tp = (const int*)d_in[2];

  char* ws = (char*)d_ws;
  bf16*   xhi   = (bf16*)(ws + XHI_OFF);
  bf16*   xlo   = (bf16*)(ws + XLO_OFF);
  float*  xn    = (float*)(ws + XN_OFF);
  float4* stats = (float4*)(ws + STATS_OFF);
  float*  acc   = (float*)(ws + ACC_OFF);

  snnl_prep<<<BATCH, 256, 0, stream>>>(x, xhi, xlo, xn, acc);
  dim3 grid(ROWTILES, COLSPLITS);
  snnl_main<<<grid, 256, 0, stream>>>(xhi, xlo, xn, y, Tp, stats);
  snnl_reduce<<<BATCH / 256, 256, 0, stream>>>(stats, acc);
  snnl_finalize<<<1, 1, 0, stream>>>(acc, (float*)d_out);
}
